// WindowEncoderSep_34634616275345
// MI455X (gfx1250) — matrix operand order repaired and checked by reading, not compiled, not run on hardware
//
#include <hip/hip_runtime.h>
#include <cstdint>

typedef __attribute__((ext_vector_type(16))) _Float16 v16h;
typedef __attribute__((ext_vector_type(8)))  _Float16 v8h;
typedef __attribute__((ext_vector_type(8)))  float    v8f;
using f16 = _Float16;

// ---------------------------------------------------------------------------
// WMMA helpers (V_WMMA_F32_16X16X32_F16), wave32 fragment layouts per ISA 7.12.2
// ---------------------------------------------------------------------------
__device__ __forceinline__ v8f wmma32(v16h a, v16h b, v8f c) {
  return __builtin_amdgcn_wmma_f32_16x16x32_f16(false, a, false, b, (short)0, c, false, false);
}

// A fragment: 16x32 (MxK), row-major f16 source with stride lda (lda % 8 == 0,
// src 16B aligned). lanes 0-15: M=lane, K={0..7,16..23}; lanes 16-31: K={8..15,24..31}
__device__ __forceinline__ v16h load_a_frag(const f16* src, int lda) {
  int lane = threadIdx.x & 31;
  int m = lane & 15;
  int hsel = lane >> 4;
  const f16* r = src + (size_t)m * lda + hsel * 8;
  v8h lo = *(const v8h*)r;
  v8h hi = *(const v8h*)(r + 16);
  return __builtin_shufflevector(lo, hi, 0, 1, 2, 3, 4, 5, 6, 7,
                                 8, 9, 10, 11, 12, 13, 14, 15);
}

// B fragment from B^T (N-major) storage: srcT[n][k], stride ldk (K dim).
// lanes 0-15: N=lane, K={0..7,16..23}; lanes 16-31: N=lane-16, K={8..15,24..31}
__device__ __forceinline__ v16h load_bT_frag(const f16* srcT, int ldk) {
  int lane = threadIdx.x & 31;
  int n = lane & 15;
  int koff = (lane >> 4) * 8;
  const f16* r = srcT + (size_t)n * ldk + koff;
  v8h lo = *(const v8h*)r;
  v8h hi = *(const v8h*)(r + 16);
  return __builtin_shufflevector(lo, hi, 0, 1, 2, 3, 4, 5, 6, 7,
                                 8, 9, 10, 11, 12, 13, 14, 15);
}

// C/D 16x16 f32: VGPR r -> row = r + 8*(lane>=16), col = lane&15

// Coalesced f16 tile store: repack 16x16 tile through per-wave LDS scratch
// (256 halfs). Write transposed T[n][m] (contiguous per lane, 1 ds_store_b128),
// then each lane emits one 16B row-segment store (2 lanes per row).
// Safe within a wave: LDS ops are in-order; explicit s_wait_dscnt between.
__device__ __forceinline__ void store_tile_f16(f16* __restrict__ dst, int ldd,
                                               const float* vals, f16* scr) {
  int lane = threadIdx.x & 31;
  int n = lane & 15, hsel = lane >> 4;
  v8h p;
#pragma unroll
  for (int r = 0; r < 8; ++r) p[r] = (f16)vals[r];
  *(v8h*)(scr + n * 16 + hsel * 8) = p;  // T[n][m]: m = r + 8*hsel
  asm volatile("s_wait_dscnt 0" ::: "memory");
  int m = lane >> 1, hh = lane & 1;
  v8h row;
#pragma unroll
  for (int i = 0; i < 8; ++i) row[i] = scr[(hh * 8 + i) * 16 + m];
  *(v8h*)(dst + (size_t)m * ldd + hh * 8) = row;
  asm volatile("s_wait_dscnt 0" ::: "memory");  // scr reused next tile
}

// Async global->LDS staging (CDNA5)
__device__ __forceinline__ void async_copy16(uint32_t lds_byte_addr, const void* gaddr) {
  asm volatile("global_load_async_to_lds_b128 %0, %1, off"
               :: "v"(lds_byte_addr), "v"((uint64_t)(uintptr_t)gaddr) : "memory");
}
__device__ __forceinline__ void async_wait0() {
  asm volatile("s_wait_asynccnt 0" ::: "memory");
}

__device__ __forceinline__ float wave_sum(float s) {
#pragma unroll
  for (int off = 16; off > 0; off >>= 1) s += __shfl_xor(s, off, 32);
  return s;
}
__device__ __forceinline__ float wave_max(float s) {
#pragma unroll
  for (int off = 16; off > 0; off >>= 1) s = fmaxf(s, __shfl_xor(s, off, 32));
  return s;
}

// ---------------------------------------------------------------------------
// Kernel 0: weight convert fp32 -> f16 (row-major W[out][in] is already B^T)
// ---------------------------------------------------------------------------
__global__ void cvt_kernel(f16* dst, const float* __restrict__ src, int nelem) {
  int i = blockIdx.x * 256 + threadIdx.x;
  if (i < nelem) dst[i] = (f16)src[i];
}

// ---------------------------------------------------------------------------
// Kernel 1: window partition + LN1.  One block per window (512 blocks, 256 thr)
// ---------------------------------------------------------------------------
__global__ void prep_kernel(const float* __restrict__ x, const float* ln1_g,
                            const float* ln1_b, float* tS, f16* h16) {
  extern __shared__ char smem[];
  float* tl = (float*)smem;  // 256 tokens x 96 channels
  int b_ = blockIdx.x;
  int b = b_ >> 8, hb = (b_ >> 4) & 15, wb = b_ & 15;
  int tid = threadIdx.x;
  int d = tid >> 6, hh = (tid >> 3) & 7, ww = tid & 7;
  size_t xbase = (size_t)b * 96 * 65536;
  int sp = (d * 128 + hb * 8 + hh) * 128 + wb * 8 + ww;
  for (int c = 0; c < 96; ++c)
    tl[tid * 96 + c] = x[xbase + (size_t)c * 65536 + sp];
  __syncthreads();

  int lane = tid & 31, w = tid >> 5;
  for (int tt = 0; tt < 32; ++tt) {
    int tok = w * 32 + tt;
    float x0 = tl[tok * 96 + lane];
    float x1 = tl[tok * 96 + lane + 32];
    float x2 = tl[tok * 96 + lane + 64];
    float mean = wave_sum(x0 + x1 + x2) * (1.0f / 96.0f);
    float d0 = x0 - mean, d1 = x1 - mean, d2 = x2 - mean;
    float var = wave_sum(d0 * d0 + d1 * d1 + d2 * d2) * (1.0f / 96.0f);
    float inv = rsqrtf(var + 1e-5f);
    size_t g = ((size_t)b_ * 256 + tok) * 96;
    tS[g + lane] = x0; tS[g + lane + 32] = x1; tS[g + lane + 64] = x2;
    h16[g + lane]      = (f16)(d0 * inv * ln1_g[lane]      + ln1_b[lane]);
    h16[g + lane + 32] = (f16)(d1 * inv * ln1_g[lane + 32] + ln1_b[lane + 32]);
    h16[g + lane + 64] = (f16)(d2 * inv * ln1_g[lane + 64] + ln1_b[lane + 64]);
  }
}

// ---------------------------------------------------------------------------
// Kernel 2: QKV GEMM (131072x96 @ 96x288) with WMMA.
// q,k: token-major [b_][head][tok][hd] via LDS-repacked b128 stores;
// v: transposed [b_][head][hd][tok] via direct per-lane v8h stores (tile rows
// are contiguous in the transposed layout).
// ---------------------------------------------------------------------------
__global__ void qkv_kernel(const f16* __restrict__ h16, const f16* __restrict__ W,
                           const float* qkv_b, f16* q16, f16* k16, f16* vT16) {
  __shared__ f16 scr[8][256];
  int w = threadIdx.x >> 5, lane = threadIdx.x & 31;
  int m0 = blockIdx.x * 128 + w * 16;
  int n = lane & 15, hsel = lane >> 4;
  int b0 = m0 >> 8, tok0 = m0 & 255;
  v16h a0 = load_a_frag(h16 + (size_t)m0 * 96, 96);
  v16h a1 = load_a_frag(h16 + (size_t)m0 * 96 + 32, 96);
  v16h a2 = load_a_frag(h16 + (size_t)m0 * 96 + 64, 96);
  for (int nt = 0; nt < 18; ++nt) {
    const f16* Wt = W + (size_t)(nt * 16) * 96;
    v8f c = {};
    c = wmma32(a0, load_bT_frag(Wt, 96), c);
    c = wmma32(a1, load_bT_frag(Wt + 32, 96), c);
    c = wmma32(a2, load_bT_frag(Wt + 64, 96), c);
    int col = nt * 16 + n;
    float bias = qkv_b[col];
    if (nt < 6) {            // ---- q: scale, repacked coalesced store
      int head = (nt * 16) >> 5, dd0 = (nt * 16) & 31;
      float vals[8];
#pragma unroll
      for (int r = 0; r < 8; ++r) vals[r] = (c[r] + bias) * 0.17677669529663687f;
      store_tile_f16(q16 + (((size_t)b0 * 3 + head) * 256 + tok0) * 32 + dd0, 32,
                     vals, scr[w]);
    } else if (nt < 12) {    // ---- k: repacked coalesced store
      int hc0 = nt * 16 - 96;
      int head = hc0 >> 5, dd0 = hc0 & 31;
      float vals[8];
#pragma unroll
      for (int r = 0; r < 8; ++r) vals[r] = c[r] + bias;
      store_tile_f16(k16 + (((size_t)b0 * 3 + head) * 256 + tok0) * 32 + dd0, 32,
                     vals, scr[w]);
    } else {                 // ---- v (transposed dst): rows contiguous per lane
      int hc0 = nt * 16 - 192;
      int head = hc0 >> 5, dd = (hc0 & 31) + n;
      v8h p;
#pragma unroll
      for (int r = 0; r < 8; ++r) p[r] = (f16)(c[r] + bias);
      *(v8h*)(vT16 + (((size_t)b0 * 3 + head) * 32 + dd) * 256 + tok0 + 8 * hsel) = p;
    }
  }
}

// ---------------------------------------------------------------------------
// Kernel 3: attention. Block = (window, head, query-half of 128 rows).
// LDS: k(256x32 f16) | vT(32x256 f16) | S(128x256 f32) | P(128x256 f16)
// ---------------------------------------------------------------------------
__global__ void attn_kernel(const f16* __restrict__ q16, const f16* __restrict__ k16,
                            const f16* __restrict__ vT16,
                            const float* __restrict__ bias_table, f16* o16) {
  extern __shared__ char smem[];
  f16*   k_l  = (f16*)smem;                        // 16384 B (256 x 32, tok-major)
  f16*   vT_l = (f16*)(smem + 16384);              // 16384 B (32 x 256, hd-major)
  float* S    = (float*)(smem + 32768);            // 131072 B
  f16*   P    = (f16*)(smem + 32768 + 131072);     // 65536 B

  int bi = blockIdx.x;
  int win = bi / 6, rem = bi % 6, head = rem >> 1, qh = rem & 1;
  int tid = threadIdx.x, lane = tid & 31, w = tid >> 5;
  size_t hb = ((size_t)win * 3 + head) * (256 * 32);

  // stage k and vT via CDNA5 async global->LDS (64 B per thread per buffer)
  {
    const f16* gk = k16 + hb + tid * 32;
    const f16* gv = vT16 + hb + tid * 32;
    uint32_t lk = (uint32_t)(uintptr_t)(k_l + tid * 32);
    uint32_t lv = (uint32_t)(uintptr_t)(vT_l + tid * 32);
#pragma unroll
    for (int i = 0; i < 4; ++i) {
      async_copy16(lk + i * 16, gk + i * 8);
      async_copy16(lv + i * 16, gv + i * 8);
    }
    async_wait0();
  }
  __syncthreads();

  int qr0 = qh * 128 + w * 16;
  int n = lane & 15, hsel = lane >> 4;
  v16h aq = load_a_frag(q16 + hb + (size_t)qr0 * 32, 32);

  // S = q @ k^T + rel-pos bias  (K = hd = 32: one WMMA per 16x16 tile)
  for (int nt = 0; nt < 16; ++nt) {
    v8f c = {};
    c = wmma32(aq, load_bT_frag(k_l + (size_t)(nt * 16) * 32, 32), c);
    int j = nt * 16 + n;
    int jh2 = j >> 4, jw2 = j & 15;
#pragma unroll
    for (int r = 0; r < 8; ++r) {
      int qi = qr0 + r + 8 * hsel;
      int ih = qi >> 4, iw = qi & 15;
      int idx = (ih - jh2 + 15) * 31 + (iw - jw2 + 15);
      S[(w * 16 + r + 8 * hsel) * 256 + j] = c[r] + bias_table[idx * 3 + head];
    }
  }
  __syncthreads();

  // softmax over 256 keys, wave handles 16 rows, 8 values/lane
  for (int rr = 0; rr < 16; ++rr) {
    int row = w * 16 + rr;
    float vals[8];
#pragma unroll
    for (int i = 0; i < 8; ++i) vals[i] = S[row * 256 + lane + 32 * i];
    float mx = vals[0];
#pragma unroll
    for (int i = 1; i < 8; ++i) mx = fmaxf(mx, vals[i]);
    mx = wave_max(mx);
    float sum = 0.0f;
#pragma unroll
    for (int i = 0; i < 8; ++i) { float e = __expf(vals[i] - mx); vals[i] = e; sum += e; }
    sum = wave_sum(sum);
    float inv = 1.0f / sum;
#pragma unroll
    for (int i = 0; i < 8; ++i) P[row * 256 + lane + 32 * i] = (f16)(vals[i] * inv);
  }
  __syncthreads();

  // O = P @ V : K = 256 (8 WMMA steps), Nout = hd = 32 (two 16-wide tiles)
  v8f c0 = {}, c1 = {};
  int ql0 = w * 16;
#pragma unroll
  for (int kk = 0; kk < 8; ++kk) {
    v16h ap = load_a_frag(P + (size_t)ql0 * 256 + kk * 32, 256);
    c0 = wmma32(ap, load_bT_frag(vT_l + kk * 32, 256), c0);
    c1 = wmma32(ap, load_bT_frag(vT_l + 16 * 256 + kk * 32, 256), c1);
  }
#pragma unroll
  for (int r = 0; r < 8; ++r) {
    int tok = qr0 + r + 8 * hsel;
    size_t orow = ((size_t)win * 256 + tok) * 96 + head * 32;
    o16[orow + n] = (f16)c0[r];
    o16[orow + 16 + n] = (f16)c1[r];
  }
}

// ---------------------------------------------------------------------------
// Kernel 4: proj GEMM + residual + LN2 (fused).  t2 (f32) and h2 (f16) out.
// ---------------------------------------------------------------------------
__global__ void proj_kernel(const f16* __restrict__ o16, const f16* __restrict__ W,
                            const float* proj_b, const float* __restrict__ tS,
                            const float* ln2_g, const float* ln2_b,
                            float* t2, f16* h2) {
  __shared__ float strip[8][16 * 96];
  int w = threadIdx.x >> 5, lane = threadIdx.x & 31;
  int m0 = blockIdx.x * 128 + w * 16;
  int n = lane & 15, hsel = lane >> 4;
  v16h a0 = load_a_frag(o16 + (size_t)m0 * 96, 96);
  v16h a1 = load_a_frag(o16 + (size_t)m0 * 96 + 32, 96);
  v16h a2 = load_a_frag(o16 + (size_t)m0 * 96 + 64, 96);
  for (int nt = 0; nt < 6; ++nt) {
    const f16* Wt = W + (size_t)(nt * 16) * 96;
    v8f c = {};
    c = wmma32(a0, load_bT_frag(Wt, 96), c);
    c = wmma32(a1, load_bT_frag(Wt + 32, 96), c);
    c = wmma32(a2, load_bT_frag(Wt + 64, 96), c);
    int col = nt * 16 + n;
    float pb = proj_b[col];
#pragma unroll
    for (int r = 0; r < 8; ++r) {
      int lr = r + 8 * hsel;
      strip[w][lr * 96 + col] = c[r] + pb + tS[(size_t)(m0 + lr) * 96 + col];
    }
  }
  __syncthreads();
  for (int rr = 0; rr < 16; ++rr) {
    float x0 = strip[w][rr * 96 + lane];
    float x1 = strip[w][rr * 96 + lane + 32];
    float x2 = strip[w][rr * 96 + lane + 64];
    float mean = wave_sum(x0 + x1 + x2) * (1.0f / 96.0f);
    float d0 = x0 - mean, d1 = x1 - mean, d2 = x2 - mean;
    float var = wave_sum(d0 * d0 + d1 * d1 + d2 * d2) * (1.0f / 96.0f);
    float inv = rsqrtf(var + 1e-5f);
    size_t g = (size_t)(m0 + rr) * 96;
    t2[g + lane] = x0; t2[g + lane + 32] = x1; t2[g + lane + 64] = x2;  // coalesced
    h2[g + lane]      = (f16)(d0 * inv * ln2_g[lane]      + ln2_b[lane]);
    h2[g + lane + 32] = (f16)(d1 * inv * ln2_g[lane + 32] + ln2_b[lane + 32]);
    h2[g + lane + 64] = (f16)(d2 * inv * ln2_g[lane + 64] + ln2_b[lane + 64]);
  }
}

// ---------------------------------------------------------------------------
// Kernel 5: fused MLP: fc1 -> GELU (kept in LDS, 128x384 f16) -> fc2 + residual
// ---------------------------------------------------------------------------
__global__ void mlp_kernel(const f16* __restrict__ h2, const f16* __restrict__ W1,
                           const float* fc1_b, const f16* __restrict__ W2,
                           const float* fc2_b, const float* __restrict__ t2,
                           float* out) {
  extern __shared__ char smem[];
  f16* hid = (f16*)smem;  // 128 x 384
  int w = threadIdx.x >> 5, lane = threadIdx.x & 31;
  int m0 = blockIdx.x * 128 + w * 16;
  int n = lane & 15, hsel = lane >> 4;
  v16h a0 = load_a_frag(h2 + (size_t)m0 * 96, 96);
  v16h a1 = load_a_frag(h2 + (size_t)m0 * 96 + 32, 96);
  v16h a2 = load_a_frag(h2 + (size_t)m0 * 96 + 64, 96);
  for (int nt = 0; nt < 24; ++nt) {
    const f16* Wt = W1 + (size_t)(nt * 16) * 96;
    v8f c = {};
    c = wmma32(a0, load_bT_frag(Wt, 96), c);
    c = wmma32(a1, load_bT_frag(Wt + 32, 96), c);
    c = wmma32(a2, load_bT_frag(Wt + 64, 96), c);
    int col = nt * 16 + n;
    float b1 = fc1_b[col];
#pragma unroll
    for (int r = 0; r < 8; ++r) {
      int lr = r + 8 * hsel;
      float xv = c[r] + b1;
      float g = 0.5f * xv * (1.0f + erff(xv * 0.7071067811865475f));
      hid[(w * 16 + lr) * 384 + col] = (f16)g;
    }
  }
  __syncthreads();
  for (int nt = 0; nt < 6; ++nt) {
    v8f c = {};
    for (int kk = 0; kk < 12; ++kk) {
      v16h a = load_a_frag(hid + (size_t)(w * 16) * 384 + kk * 32, 384);
      c = wmma32(a, load_bT_frag(W2 + (size_t)(nt * 16) * 384 + kk * 32, 384), c);
    }
    int col = nt * 16 + n;
    float b2 = fc2_b[col];
#pragma unroll
    for (int r = 0; r < 8; ++r) {
      int gm = m0 + r + 8 * hsel;
      out[(size_t)gm * 96 + col] = c[r] + b2 + t2[(size_t)gm * 96 + col];
    }
  }
}

// ---------------------------------------------------------------------------
extern "C" void kernel_launch(void* const* d_in, const int* in_sizes, int n_in,
                              void* d_out, int out_size, void* d_ws, size_t ws_size,
                              hipStream_t stream) {
  (void)in_sizes; (void)n_in; (void)out_size; (void)ws_size;
  const float* x          = (const float*)d_in[0];
  const float* qkv_w      = (const float*)d_in[1];
  const float* qkv_b      = (const float*)d_in[2];
  const float* proj_w     = (const float*)d_in[3];
  const float* proj_b     = (const float*)d_in[4];
  const float* bias_table = (const float*)d_in[5];
  const float* ln1_g      = (const float*)d_in[6];
  const float* ln1_b      = (const float*)d_in[7];
  const float* ln2_g      = (const float*)d_in[8];
  const float* ln2_b      = (const float*)d_in[9];
  const float* fc1_w      = (const float*)d_in[10];
  const float* fc1_b      = (const float*)d_in[11];
  const float* fc2_w      = (const float*)d_in[12];
  const float* fc2_b      = (const float*)d_in[13];
  float* out = (float*)d_out;

  const size_t M = 512 * 256;
  char* ws = (char*)d_ws;
  size_t off = 0;
  auto alloc = [&](size_t bytes) {
    char* p = ws + off;
    off += (bytes + 255) & ~(size_t)255;
    return p;
  };
  f16*   qkvW16 = (f16*)alloc((size_t)288 * 96 * 2);
  f16*   projW16 = (f16*)alloc((size_t)96 * 96 * 2);
  f16*   fc1W16 = (f16*)alloc((size_t)384 * 96 * 2);
  f16*   fc2W16 = (f16*)alloc((size_t)96 * 384 * 2);
  float* tS   = (float*)alloc(M * 96 * 4);
  f16*   h16  = (f16*)alloc(M * 96 * 2);   // reused as o16
  f16*   q16  = (f16*)alloc(M * 96 * 2);   // reused as h2
  f16*   k16  = (f16*)alloc(M * 96 * 2);
  f16*   vT16 = (f16*)alloc(M * 96 * 2);
  float* t2   = (float*)alloc(M * 96 * 4);
  f16* o16 = h16;
  f16* h2  = q16;

  cvt_kernel<<<(288 * 96 + 255) / 256, 256, 0, stream>>>(qkvW16, qkv_w, 288 * 96);
  cvt_kernel<<<(96 * 96 + 255) / 256, 256, 0, stream>>>(projW16, proj_w, 96 * 96);
  cvt_kernel<<<(384 * 96 + 255) / 256, 256, 0, stream>>>(fc1W16, fc1_w, 384 * 96);
  cvt_kernel<<<(96 * 384 + 255) / 256, 256, 0, stream>>>(fc2W16, fc2_w, 96 * 384);

  prep_kernel<<<512, 256, 256 * 96 * 4, stream>>>(x, ln1_g, ln1_b, tS, h16);
  qkv_kernel<<<1024, 256, 0, stream>>>(h16, qkvW16, qkv_b, q16, k16, vT16);
  attn_kernel<<<3072, 256, 229376, stream>>>(q16, k16, vT16, bias_table, o16);
  proj_kernel<<<1024, 256, 0, stream>>>(o16, projW16, proj_b, tS, ln2_g, ln2_b, t2, h2);
  mlp_kernel<<<1024, 256, 128 * 384 * 2, stream>>>(h2, fc1W16, fc1_b, fc2W16, fc2_b, t2, out);
}
